// CosClassifier_21380347200230
// MI455X (gfx1250) — matrix operand order repaired and checked by reading, not compiled, not run on hardware
//
#include <hip/hip_runtime.h>
#include <hip/hip_bf16.h>

typedef _Float16 half4v __attribute__((ext_vector_type(4)));
typedef _Float16 half8v __attribute__((ext_vector_type(8)));
typedef _Float16 v16h   __attribute__((ext_vector_type(16)));
typedef float    v8f    __attribute__((ext_vector_type(8)));
typedef float    f4v    __attribute__((ext_vector_type(4)));

#define D_K    256   // inner dimension
#define LDC    4096  // output row stride (C)
#define KC     64    // K chunk staged in LDS
#define LDKP   72    // padded halves per LDS row (64 + 8 pad -> conflict-free-ish)

// -------- per-row L2 norms: one wave32 per row, D=256 --------
__global__ __launch_bounds__(256) void row_norms_256(const float* __restrict__ src,
                                                     float* __restrict__ dst) {
  const int lane = threadIdx.x & 31;
  const int wave = threadIdx.x >> 5;
  const int row  = blockIdx.x * 8 + wave;
  const float* p = src + (size_t)row * D_K;
  float s = 0.f;
#pragma unroll
  for (int j = 0; j < 8; ++j) {
    float v = p[j * 32 + lane];
    s += v * v;
  }
#pragma unroll
  for (int off = 16; off > 0; off >>= 1)
    s += __shfl_xor(s, off, 32);
  if (lane == 0) dst[row] = sqrtf(s);
}

// -------- split-f16 WMMA GEMM + cosine epilogue --------
// out[b,c] = (X[b,:]·P[c,:]) * s*10 / max(xn[b]*pn[c], 1e-8)
__global__ __launch_bounds__(256) void cos_gemm(const float* __restrict__ X,
                                                const float* __restrict__ P,
                                                const float* __restrict__ scal,
                                                const float* __restrict__ xn,
                                                const float* __restrict__ pn,
                                                float* __restrict__ out) {
  __shared__ __align__(16) _Float16 Ah[128 * LDKP];
  __shared__ __align__(16) _Float16 Al[128 * LDKP];
  __shared__ __align__(16) _Float16 Bh[128 * LDKP];
  __shared__ __align__(16) _Float16 Bl[128 * LDKP];

  const int t    = threadIdx.x;
  const int lane = t & 31;
  const int wave = t >> 5;
  const int wm   = wave >> 1;   // 0..3  (rows:  32 each)
  const int wn   = wave & 1;    // 0..1  (cols:  64 each)
  const int bm   = blockIdx.y * 128;
  const int bn   = blockIdx.x * 128;

  const v8f vzero = {0.f, 0.f, 0.f, 0.f, 0.f, 0.f, 0.f, 0.f};
  v8f acc[2][4];
#pragma unroll
  for (int i = 0; i < 2; ++i)
#pragma unroll
    for (int j = 0; j < 4; ++j) acc[i][j] = vzero;

  for (int kc = 0; kc < D_K; kc += KC) {
    __syncthreads();  // previous-iteration readers done before we overwrite LDS
    // ---- stage 128x64 fp32 of X and P, split into (hi,lo) f16 in LDS ----
#pragma unroll
    for (int i = 0; i < 8; ++i) {
      int f4 = t + i * 256;        // 0..2047 float4-slots over 128x16
      int r  = f4 >> 4;            // row in tile
      int c  = (f4 & 15) << 2;     // col (floats) in chunk
      f4v a = *(const f4v*)(X + (size_t)(bm + r) * D_K + kc + c);
      f4v b = *(const f4v*)(P + (size_t)(bn + r) * D_K + kc + c);
      half4v ah, al, bh, bl;
#pragma unroll
      for (int j = 0; j < 4; ++j) {
        _Float16 h = (_Float16)a[j];
        ah[j] = h;
        al[j] = (_Float16)(a[j] - (float)h);
        h = (_Float16)b[j];
        bh[j] = h;
        bl[j] = (_Float16)(b[j] - (float)h);
      }
      *(half4v*)&Ah[r * LDKP + c] = ah;
      *(half4v*)&Al[r * LDKP + c] = al;
      *(half4v*)&Bh[r * LDKP + c] = bh;
      *(half4v*)&Bl[r * LDKP + c] = bl;
    }
    __syncthreads();

    // ---- 2 K-steps of 32 per chunk ----
#pragma unroll
    for (int k0 = 0; k0 < KC; k0 += 32) {
      // A fragments (16x32 f16 layout: lanes 0-15 start K+0, lanes 16-31 start K+8;
      // elems 0..7 = K..K+7, elems 8..15 = K+16..K+23)
      v16h a_hi[2], a_lo[2];
      const int arow_lo = lane & 15;
      const int ako     = k0 + ((lane >> 4) << 3);
#pragma unroll
      for (int ms = 0; ms < 2; ++ms) {
        int row = wm * 32 + ms * 16 + arow_lo;
        const _Float16* ph = &Ah[row * LDKP + ako];
        const _Float16* pl = &Al[row * LDKP + ako];
        half8v h0 = *(const half8v*)ph;
        half8v h1 = *(const half8v*)(ph + 16);
        half8v l0 = *(const half8v*)pl;
        half8v l1 = *(const half8v*)(pl + 16);
        a_hi[ms] = __builtin_shufflevector(h0, h1, 0,1,2,3,4,5,6,7,8,9,10,11,12,13,14,15);
        a_lo[ms] = __builtin_shufflevector(l0, l1, 0,1,2,3,4,5,6,7,8,9,10,11,12,13,14,15);
      }
      // B fragments (32x16 f16 layout: lane holds one column (= one proto row),
      // lanes 0-15 start K+0, lanes 16-31 start K+8; elems 0..7 = K..K+7, elems 8..15 = K+16..K+23)
      const int bko = k0 + ((lane >> 4) << 3);
#pragma unroll
      for (int ns = 0; ns < 4; ++ns) {
        int col = wn * 64 + ns * 16 + (lane & 15);
        const _Float16* ph = &Bh[col * LDKP + bko];
        const _Float16* pl = &Bl[col * LDKP + bko];
        half8v h0 = *(const half8v*)ph;
        half8v h1 = *(const half8v*)(ph + 16);
        half8v l0 = *(const half8v*)pl;
        half8v l1 = *(const half8v*)(pl + 16);
        v16h b_hi = __builtin_shufflevector(h0, h1, 0,1,2,3,4,5,6,7,8,9,10,11,12,13,14,15);
        v16h b_lo = __builtin_shufflevector(l0, l1, 0,1,2,3,4,5,6,7,8,9,10,11,12,13,14,15);
#pragma unroll
        for (int ms = 0; ms < 2; ++ms) {
          acc[ms][ns] = __builtin_amdgcn_wmma_f32_16x16x32_f16(
              false, a_hi[ms], false, b_hi, (short)0, acc[ms][ns], false, false);
          acc[ms][ns] = __builtin_amdgcn_wmma_f32_16x16x32_f16(
              false, a_hi[ms], false, b_lo, (short)0, acc[ms][ns], false, false);
          acc[ms][ns] = __builtin_amdgcn_wmma_f32_16x16x32_f16(
              false, a_lo[ms], false, b_hi, (short)0, acc[ms][ns], false, false);
        }
      }
    }
  }

  // ---- epilogue: C/D layout -> VGPR v: lanes 0-15 row v, lanes 16-31 row v+8 ----
  const float s10 = scal[0] * 10.0f;
#pragma unroll
  for (int ms = 0; ms < 2; ++ms) {
    int row0 = bm + wm * 32 + ms * 16 + ((lane >> 4) << 3);
    float xnv[8];
#pragma unroll
    for (int v = 0; v < 8; ++v) xnv[v] = xn[row0 + v];
#pragma unroll
    for (int ns = 0; ns < 4; ++ns) {
      int col  = bn + wn * 64 + ns * 16 + (lane & 15);
      float pv = pn[col];
#pragma unroll
      for (int v = 0; v < 8; ++v) {
        float denom = fmaxf(xnv[v] * pv, 1e-8f);
        out[(size_t)(row0 + v) * LDC + col] = acc[ms][ns][v] * s10 / denom;
      }
    }
  }
}

extern "C" void kernel_launch(void* const* d_in, const int* in_sizes, int n_in,
                              void* d_out, int out_size, void* d_ws, size_t ws_size,
                              hipStream_t stream) {
  const float* X    = (const float*)d_in[0];  // (16384, 256)
  const float* P    = (const float*)d_in[1];  // (4096, 256)
  const float* scal = (const float*)d_in[2];  // (1,)
  float* out = (float*)d_out;                 // (16384, 4096)

  const int B = 16384, C = 4096;
  float* xn = (float*)d_ws;          // B floats
  float* pn = xn + B;                // C floats

  row_norms_256<<<B / 8, 256, 0, stream>>>(X, xn);
  row_norms_256<<<C / 8, 256, 0, stream>>>(P, pn);

  dim3 grid(C / 128, B / 128);       // (32, 128)
  cos_gemm<<<grid, 256, 0, stream>>>(X, P, scal, xn, pn, out);
}
